// MultiHeadAttention_7550552507120
// MI455X (gfx1250) — matrix operand order repaired and checked by reading, not compiled, not run on hardware
//
#include <hip/hip_runtime.h>
#include <hip/hip_bf16.h>
#include <stdint.h>

// MHA forward for B=4,S=2048,E=512,H=8,DK=64 on gfx1250 (wave32, WMMA).
// Pipeline: (0) one-shot f32->f16 conversion of activations+weights,
//           (1) QKV projections (32x64 tiles/wave, fully unrolled K loop ->
//               immediate-offset b128 loads, compiler-pipelined),
//           (2) flash-attention with transposed score tiles + online softmax,
//           (3) output projection -> fp32 d_out.

#define BB_ 4
#define SS_ 2048
#define EE_ 512
#define HH_ 8
#define DKK 64
#define NEGV (-1000000000.0f)

typedef __attribute__((ext_vector_type(16))) _Float16 v16h;
typedef __attribute__((ext_vector_type(8)))  float    v8f;

union U16x { v16h v; uint32_t u[8]; _Float16 h[16]; };
union Pack8h { _Float16 h[8]; uint4 u4; };

// A-matrix (16-bit, 16x32 MxK) element K-offset for (vgpr v, lane-half h)
static __device__ __forceinline__ int koffA(int v, int h) {
    return ((v < 4) ? 0 : 16) + h * 8 + (v & 3) * 2;
}
// B-matrix (16-bit, 32x16 KxN) element K-offset for (vgpr v, lane-half h)
static __device__ __forceinline__ int koffB(int v, int h) {
    return ((v < 4) ? 0 : 16) + h * 8 + (v & 3) * 2;
}

static __device__ __forceinline__ v8f wmma_f16(const v16h& a, const v16h& b, const v8f& c) {
    return __builtin_amdgcn_wmma_f32_16x16x32_f16(false, a, false, b, (short)0, c, false, false);
}

// ---------------------------------------------------------------------------
// Phase 0: elementwise f32 -> f16, 8 elements per thread.
// ---------------------------------------------------------------------------
__global__ void __launch_bounds__(256)
cvt_f16_kernel(const float* __restrict__ src, _Float16* __restrict__ dst, int n8)
{
    const int i = blockIdx.x * 256 + threadIdx.x;
    if (i >= n8) return;
    const float4* s = (const float4*)src;
    const float4 x0 = s[2 * i];
    const float4 x1 = s[2 * i + 1];
    Pack8h o;
    o.h[0] = (_Float16)x0.x; o.h[1] = (_Float16)x0.y;
    o.h[2] = (_Float16)x0.z; o.h[3] = (_Float16)x0.w;
    o.h[4] = (_Float16)x1.x; o.h[5] = (_Float16)x1.y;
    o.h[6] = (_Float16)x1.z; o.h[7] = (_Float16)x1.w;
    ((uint4*)dst)[i] = o.u4;
}

// ---------------------------------------------------------------------------
// Phase 1: Y = X @ W^T + bias  (X f16 [B*S,E], W f16 [E,E], bias f32)
// mode 0: write f16 dst as [b,h,s,d]   (Q, K)
// mode 2: write f16 dst as [b,h,d,s]   (V transposed; packed b128 stores)
// Each wave: one 32x64 output tile; K loop fully unrolled (immediate offsets).
// grid.x = (8192/32)*(512/64)/4 = 512 blocks of 128 threads.
// ---------------------------------------------------------------------------
__global__ void __launch_bounds__(128)
proj_kernel(const _Float16* __restrict__ X, const _Float16* __restrict__ W,
            const float* __restrict__ bias, _Float16* __restrict__ dst, int mode)
{
    const int lane = threadIdx.x & 31;
    const int wave = threadIdx.x >> 5;
    const int l16  = lane & 15;
    const int h2   = lane >> 4;
    const int gw   = blockIdx.x * 4 + wave;
    const int mt   = gw >> 3;             // 256 row tiles of 32
    const int nbase = (gw & 7) * 64;      // 8 col groups of 64

    // per-lane base pointers (u32 units) -> all inner loads are base + imm
    const uint32_t* Xu0 = (const uint32_t*)X + (((size_t)(mt * 32 + l16)) * EE_ >> 1);
    const uint32_t* Xu1 = Xu0 + (16 * EE_ >> 1);
    const uint32_t* Wj0 = (const uint32_t*)W + (((size_t)(nbase + l16)) * EE_ >> 1);

    v8f acc0[4] = {}, acc1[4] = {};
#pragma unroll
    for (int kb = 0; kb < EE_; kb += 32) {
        U16x a0, a1;
#pragma unroll
        for (int v = 0; v < 8; ++v) {
            const int o = (kb + koffA(v, h2)) >> 1;
            a0.u[v] = Xu0[o];
            a1.u[v] = Xu1[o];
        }
#pragma unroll
        for (int j = 0; j < 4; ++j) {
            U16x b;
#pragma unroll
            for (int v = 0; v < 8; ++v)
                b.u[v] = Wj0[j * (16 * EE_ / 2) + ((kb + koffB(v, h2)) >> 1)];
            acc0[j] = wmma_f16(a0.v, b.v, acc0[j]);
            acc1[j] = wmma_f16(a1.v, b.v, acc1[j]);
        }
    }

#pragma unroll
    for (int j = 0; j < 4; ++j) {
        const int n  = nbase + j * 16 + l16;   // output feature (lane's column)
        const float bn = bias[n];
        const int hh = n >> 6;                 // head
        const int d  = n & (DKK - 1);          // dim within head
#pragma unroll
        for (int r = 0; r < 2; ++r) {
            const v8f& acc = r ? acc1[j] : acc0[j];
            const int m0 = mt * 32 + r * 16 + 8 * h2;  // 8-aligned row base
            const int bb = m0 >> 11;                   // /S (no crossing: 8-aligned)
            const int s0 = m0 & (SS_ - 1);
            if (mode == 2) {
                Pack8h o;
#pragma unroll
                for (int v = 0; v < 8; ++v) o.h[v] = (_Float16)(acc[v] + bn);
                *(uint4*)&dst[(((size_t)(bb * HH_ + hh)) * DKK + d) * SS_ + s0] = o.u4;
            } else {
#pragma unroll
                for (int v = 0; v < 8; ++v)
                    dst[(((size_t)(bb * HH_ + hh)) * SS_ + s0 + v) * DKK + d] =
                        (_Float16)(acc[v] + bn);
            }
        }
    }
}

// ---------------------------------------------------------------------------
// Phase 2: flash attention. One wave per 16-query tile; loop keys in blocks
// of 32. Score tiles computed TRANSPOSED (S^T = K * Q^T) so each lane owns a
// query column -> softmax over keys is register-local + shfl_xor(16).
// Q kept resident in B-matrix registers; V^T loads feed the A side of PV.
// grid = (32, 32): x = 128 q-tiles / 4 waves, y = B*H.
// ---------------------------------------------------------------------------
__global__ void __launch_bounds__(128)
flash_attn_kernel(const _Float16* __restrict__ Qw, const _Float16* __restrict__ Kw,
                  const _Float16* __restrict__ Vt, const int* __restrict__ mask,
                  _Float16* __restrict__ ctx)
{
    __shared__ _Float16 plds[4][32 * 16];   // per-wave P^T staging (D->B relayout)

    const int lane = threadIdx.x & 31;
    const int wave = threadIdx.x >> 5;
    const int l16  = lane & 15;
    const int h2   = lane >> 4;
    const int bh   = blockIdx.y;
    const int bb   = bh >> 3;
    const int hh   = bh & 7;
    const int q    = blockIdx.x * 64 + wave * 16 + l16;   // this lane's query

    const size_t headQK = (size_t)(bb * HH_ + hh) * SS_ * DKK;  // elem base [b,h,s,d]
    const size_t headV  = (size_t)(bb * HH_ + hh) * DKK * SS_;  // elem base [b,h,d,s]

    // per-lane strength-reduced base pointers (u32 units)
    const uint32_t* QuL = (const uint32_t*)Qw + ((headQK + (size_t)q * DKK) >> 1);
    const uint32_t* KuL = (const uint32_t*)Kw + ((headQK + (size_t)l16 * DKK) >> 1);
    const uint32_t* VuL = (const uint32_t*)Vt + ((headV + (size_t)l16 * SS_) >> 1);
    const int* mrow = mask + ((long)bb * SS_ + q) * SS_;

    // Q^T resident in B-layout registers (dims 0-31 and 32-63)
    U16x bq0, bq1;
#pragma unroll
    for (int v = 0; v < 8; ++v) {
        const int kk = koffB(v, h2) >> 1;
        bq0.u[v] = QuL[kk];
        bq1.u[v] = QuL[kk + 16];
    }

    float mrun = -3.0e38f, lrun = 0.0f;
    v8f O[4] = {};

#pragma unroll 2
    for (int kb = 0; kb < SS_; kb += 32) {
        // prefetch next key block into caches (global_prefetch)
        if (kb + 32 < SS_) {
            __builtin_prefetch((const void*)(KuL + (kb + 32) * 32), 0, 1);
            __builtin_prefetch((const void*)(KuL + (kb + 48) * 32), 0, 1);
            __builtin_prefetch((const void*)(VuL + lane * 1024 + ((kb + 32) >> 1)), 0, 1);
            __builtin_prefetch((const void*)(mrow + kb + 32), 0, 1);
        }

        // ---- S^T tiles: D0 = keys kb..kb+15, D1 = keys kb+16..kb+31 ----
        v8f d0 = {}, d1 = {};
        {
            U16x ak;
#pragma unroll
            for (int v = 0; v < 8; ++v) ak.u[v] = KuL[kb * 32 + (koffA(v, h2) >> 1)];
            d0 = wmma_f16(ak.v, bq0.v, d0);
#pragma unroll
            for (int v = 0; v < 8; ++v) ak.u[v] = KuL[kb * 32 + 16 + (koffA(v, h2) >> 1)];
            d0 = wmma_f16(ak.v, bq1.v, d0);
        }
        {
            U16x ak;
#pragma unroll
            for (int v = 0; v < 8; ++v) ak.u[v] = KuL[kb * 32 + 512 + (koffA(v, h2) >> 1)];
            d1 = wmma_f16(ak.v, bq0.v, d1);
#pragma unroll
            for (int v = 0; v < 8; ++v) ak.u[v] = KuL[kb * 32 + 528 + (koffA(v, h2) >> 1)];
            d1 = wmma_f16(ak.v, bq1.v, d1);
        }

        // ---- mask + (*sqrt(dk)=8 quirk), block max over this lane's 16 keys
        float s0[8], s1[8];
        float bm = -3.0e38f;
#pragma unroll
        for (int v = 0; v < 8; ++v) {
            const int k0 = kb + v + 8 * h2;        // key held in d0[v]
            const float a0 = mrow[k0]      ? d0[v] * 8.0f : NEGV;
            const float a1 = mrow[k0 + 16] ? d1[v] * 8.0f : NEGV;
            s0[v] = a0; s1[v] = a1;
            bm = fmaxf(bm, fmaxf(a0, a1));
        }
        bm = fmaxf(bm, __shfl_xor(bm, 16, 32));    // partner lane holds same query
        const float mnew  = fmaxf(mrun, bm);
        const float scale = __expf(mrun - mnew);

        // ---- exp, running sum, stage P^T (f16) into LDS in D layout ----
        float psum = 0.0f;
#pragma unroll
        for (int v = 0; v < 8; ++v) {
            const float p0 = __expf(s0[v] - mnew);
            const float p1 = __expf(s1[v] - mnew);
            psum += p0 + p1;
            plds[wave][(v + 8 * h2) * 16 + l16]      = (_Float16)p0;
            plds[wave][(16 + v + 8 * h2) * 16 + l16] = (_Float16)p1;
        }
        psum += __shfl_xor(psum, 16, 32);
        lrun = lrun * scale + psum;
        mrun = mnew;
#pragma unroll
        for (int t = 0; t < 4; ++t)
#pragma unroll
            for (int v = 0; v < 8; ++v) O[t][v] *= scale;

        // ---- reload P^T in B-matrix layout (same wave: DS pipe is in-order)
        U16x bp;
#pragma unroll
        for (int v = 0; v < 8; ++v) {
            const int kk = koffB(v, h2);
            bp.h[2 * v]     = plds[wave][kk * 16 + l16];
            bp.h[2 * v + 1] = plds[wave][(kk + 1) * 16 + l16];
        }

        // ---- O^T += V^T * P^T  (4 dim tiles of 16) ----
#pragma unroll
        for (int t = 0; t < 4; ++t) {
            U16x av;
#pragma unroll
            for (int v = 0; v < 8; ++v)
                av.u[v] = VuL[t * 16384 + ((kb + koffA(v, h2)) >> 1)];
            O[t] = wmma_f16(av.v, bp.v, O[t]);
        }
    }

    // ---- normalize, write context in [b,h,s,d] order (reference's reshape
    //      quirk); 8 contiguous halves per lane -> one b128 store per tile
    const float inv = 1.0f / lrun;
#pragma unroll
    for (int t = 0; t < 4; ++t) {
        Pack8h o;
#pragma unroll
        for (int v = 0; v < 8; ++v) o.h[v] = (_Float16)(O[t][v] * inv);
        *(uint4*)&ctx[headQK + (size_t)q * DKK + 16 * t + 8 * h2] = o.u4;
    }
}

// ---------------------------------------------------------------------------
// Phase 3: out = ctx @ Wo^T + bo  (ctx/Wo f16, out fp32). 32x64 tile/wave,
// fully unrolled K loop. grid.x = 512.
// ---------------------------------------------------------------------------
__global__ void __launch_bounds__(128)
out_proj_kernel(const _Float16* __restrict__ ctx, const _Float16* __restrict__ Wo,
                const float* __restrict__ bo, float* __restrict__ out)
{
    const int lane = threadIdx.x & 31;
    const int wave = threadIdx.x >> 5;
    const int l16  = lane & 15;
    const int h2   = lane >> 4;
    const int gw   = blockIdx.x * 4 + wave;
    const int mt   = gw >> 3;
    const int nbase = (gw & 7) * 64;

    const uint32_t* Cu0 = (const uint32_t*)ctx + (((size_t)(mt * 32 + l16)) * EE_ >> 1);
    const uint32_t* Cu1 = Cu0 + (16 * EE_ >> 1);
    const uint32_t* Wj0 = (const uint32_t*)Wo + (((size_t)(nbase + l16)) * EE_ >> 1);

    v8f acc0[4] = {}, acc1[4] = {};
#pragma unroll
    for (int kb = 0; kb < EE_; kb += 32) {
        U16x a0, a1;
#pragma unroll
        for (int v = 0; v < 8; ++v) {
            const int o = (kb + koffA(v, h2)) >> 1;
            a0.u[v] = Cu0[o];
            a1.u[v] = Cu1[o];
        }
#pragma unroll
        for (int j = 0; j < 4; ++j) {
            U16x b;
#pragma unroll
            for (int v = 0; v < 8; ++v)
                b.u[v] = Wj0[j * (16 * EE_ / 2) + ((kb + koffB(v, h2)) >> 1)];
            acc0[j] = wmma_f16(a0.v, b.v, acc0[j]);
            acc1[j] = wmma_f16(a1.v, b.v, acc1[j]);
        }
    }

#pragma unroll
    for (int j = 0; j < 4; ++j) {
        const int n  = nbase + j * 16 + l16;
        const float bn = bo[n];
#pragma unroll
        for (int r = 0; r < 2; ++r) {
            const v8f& acc = r ? acc1[j] : acc0[j];
#pragma unroll
            for (int v = 0; v < 8; ++v) {
                const int m = mt * 32 + r * 16 + v + 8 * h2;
                out[(size_t)m * EE_ + n] = acc[v] + bn;
            }
        }
    }
}

extern "C" void kernel_launch(void* const* d_in, const int* in_sizes, int n_in,
                              void* d_out, int out_size, void* d_ws, size_t ws_size,
                              hipStream_t stream) {
    const float* query  = (const float*)d_in[0];
    const float* key_in = (const float*)d_in[1];
    const float* value  = (const float*)d_in[2];
    const int*   mask   = (const int*)  d_in[3];
    const float* Wq = (const float*)d_in[4];
    const float* bq = (const float*)d_in[5];
    const float* Wk = (const float*)d_in[6];
    const float* bk = (const float*)d_in[7];
    const float* Wv = (const float*)d_in[8];
    const float* bv = (const float*)d_in[9];
    const float* Wo = (const float*)d_in[10];
    const float* bo = (const float*)d_in[11];
    float* out = (float*)d_out;

    const size_t N1 = (size_t)BB_ * SS_ * EE_;   // 4,194,304 elements
    const size_t NW = (size_t)EE_ * EE_;         // 262,144 elements
    _Float16* base = (_Float16*)d_ws;
    _Float16* xq  = base;                 // f16 activations
    _Float16* xk  = xq  + N1;
    _Float16* xv  = xk  + N1;
    _Float16* qws = xv  + N1;             // projected Q [b,h,s,d]
    _Float16* kws = qws + N1;             // projected K [b,h,s,d]
    _Float16* vws = kws + N1;             // projected V transposed [b,h,d,s]
    _Float16* ctx = vws + N1;             // attention output [b,h,s,d]
    _Float16* wqh = ctx + N1;             // f16 weights
    _Float16* wkh = wqh + NW;
    _Float16* wvh = wkh + NW;
    _Float16* woh = wvh + NW;

    dim3 blk(128);
    const int gCvtA = (int)(N1 / 8 / 256);   // 2048 blocks
    const int gCvtW = (int)(NW / 8 / 256);   // 128 blocks
    cvt_f16_kernel<<<gCvtA, 256, 0, stream>>>(query,  xq, (int)(N1 / 8));
    cvt_f16_kernel<<<gCvtA, 256, 0, stream>>>(key_in, xk, (int)(N1 / 8));
    cvt_f16_kernel<<<gCvtA, 256, 0, stream>>>(value,  xv, (int)(N1 / 8));
    cvt_f16_kernel<<<gCvtW, 256, 0, stream>>>(Wq, wqh, (int)(NW / 8));
    cvt_f16_kernel<<<gCvtW, 256, 0, stream>>>(Wk, wkh, (int)(NW / 8));
    cvt_f16_kernel<<<gCvtW, 256, 0, stream>>>(Wv, wvh, (int)(NW / 8));
    cvt_f16_kernel<<<gCvtW, 256, 0, stream>>>(Wo, woh, (int)(NW / 8));

    dim3 gProj(512);                 // (8192/32)*(512/64)/4 waves
    proj_kernel<<<gProj, blk, 0, stream>>>(xq, wqh, bq, qws, 0);
    proj_kernel<<<gProj, blk, 0, stream>>>(xk, wkh, bk, kws, 0);
    proj_kernel<<<gProj, blk, 0, stream>>>(xv, wvh, bv, vws, 2);

    flash_attn_kernel<<<dim3(32, 32), blk, 0, stream>>>(qws, kws, vws, mask, ctx);

    out_proj_kernel<<<gProj, blk, 0, stream>>>(ctx, woh, bo, out);
}
